// MambaBlock_69861938037096
// MI455X (gfx1250) — matrix operand order repaired and checked by reading, not compiled, not run on hardware
//
#include <hip/hip_runtime.h>

typedef __attribute__((ext_vector_type(16))) _Float16 v16h;
typedef __attribute__((ext_vector_type(8)))  float    v8f;
typedef __attribute__((ext_vector_type(4)))  float    f4;
typedef __attribute__((ext_vector_type(4)))  _Float16 h4;
typedef __attribute__((ext_vector_type(4)))  unsigned v4u;
typedef __attribute__((ext_vector_type(8)))  int      v8i;
typedef __attribute__((ext_vector_type(4)))  int      v4i;

#define BQ   4
#define LQ   2048
#define DMQ  1024
#define DIQ  2048
#define DSQ  16
#define DTRQ 64

#if defined(__has_builtin)
#  if __has_builtin(__builtin_amdgcn_tensor_load_to_lds) && \
      __has_builtin(__builtin_amdgcn_s_wait_tensorcnt)
#    define HAVE_TDM 1
#  endif
#endif
#ifndef HAVE_TDM
#  define HAVE_TDM 0
#endif

#if HAVE_TDM
// Issue a 2-D TDM descriptor: tile tw(x) x th(y) of 4-byte elements,
// row stride ld (elements), global source gsrc, LDS dest lds_off (bytes).
__device__ __forceinline__ void tdm_load_2d(const float* gsrc, unsigned lds_off,
                                            int ld, int tw, int th, int tdim1)
{
  unsigned long long ga = (unsigned long long)(const void*)gsrc;
  v4u g0;
  g0[0] = 1u;                                    // count=1, user mode
  g0[1] = lds_off;                               // lds_addr (bytes)
  g0[2] = (unsigned)ga;                          // global_addr[31:0]
  g0[3] = (unsigned)((ga >> 32) & 0x01FFFFFFu)   // global_addr[56:32]
          | 0x80000000u;                         // type=2 ("image")
  v8i g1;
  g1[0] = 0x20000;                               // data_size=4B, no multicast
  g1[1] = (int)(((unsigned)ld & 0xFFFFu) << 16);            // tensor_dim0 lo16
  g1[2] = (int)((((unsigned)ld >> 16) & 0xFFFFu)
                | ((unsigned)tdim1 << 16));      // dim0 hi16 | tensor_dim1 lo16
  g1[3] = (int)((unsigned)tw << 16);             // tensor_dim1 hi | tile_dim0
  g1[4] = th;                                    // tile_dim1
  g1[5] = ld;                                    // tensor_dim0_stride lo32
  g1[6] = 0;
  g1[7] = 0;
  v4i gz = {0, 0, 0, 0};
#  if __clang_major__ >= 23
  v8i z8 = {0, 0, 0, 0, 0, 0, 0, 0};
  __builtin_amdgcn_tensor_load_to_lds(g0, g1, gz, gz, z8, 0);
#  else
  __builtin_amdgcn_tensor_load_to_lds(g0, g1, gz, gz, 0);
#  endif
}
#endif

// ---------------------------------------------------------------------------
// Tiled WMMA GEMM: C = epilogue( A(MxK, row-major, ld lda) * W(KxN, row-major) )
// Block = 256 threads = 8 waves (2 M x 4 N), tile 64x128, K-step 32.
// GUARD=false: fully tiled fast path; USE_TDM stages both tiles via the
//              Tensor Data Mover (TENSORcnt), else batched global_load_b128.
// GUARD=true : ragged-N path (clamp+select, no EXEC branches).
// epilogue==1: softplus(v + bias[n]).
// ---------------------------------------------------------------------------
template<bool GUARD, bool USE_TDM>
__global__ __launch_bounds__(256) void wmma_gemm_kernel(
    const float* __restrict__ A, const float* __restrict__ W,
    float* __restrict__ C, int M, int N, int K, int lda,
    const float* __restrict__ bias, int epilogue)
{
  __shared__ _Float16 sA[64 * 32];    // [m][k] f16
  __shared__ _Float16 sB[128 * 34];   // [n][k] f16 (transposed, padded)
#if HAVE_TDM
  __shared__ float sAraw[USE_TDM ? 64 * 32 : 1];    // TDM landing zones (f32)
  __shared__ float sBraw[USE_TDM ? 32 * 128 : 1];
#endif

  const int tid  = threadIdx.x;
  const int lane = tid & 31;
  const int wv   = tid >> 5;
  const int wm   = wv >> 2;   // 0..1
  const int wn   = wv & 3;    // 0..3
  const int g    = lane >> 4; // lane half
  const int lr   = lane & 15;

  const int m_blk = blockIdx.y * 64;
  const int n_blk = blockIdx.x * 128;

  const v8f vzero = {0.f, 0.f, 0.f, 0.f, 0.f, 0.f, 0.f, 0.f};
  v8f acc[2][2];
#pragma unroll
  for (int i = 0; i < 2; ++i)
#pragma unroll
    for (int j = 0; j < 2; ++j) acc[i][j] = vzero;

  // Per-thread running pointers (no per-iteration 64-bit multiplies).
  const size_t lda32 = (size_t)32 * lda;   // A: 32 rows
  const size_t n8    = (size_t)8 * N;      // W: 8 rows
  const float* ap = A + (size_t)(m_blk + (tid >> 3)) * lda + (tid & 7) * 4;
  const float* wp = W + (size_t)(tid >> 5) * N + n_blk + (tid & 31) * 4;

  for (int k0 = 0; k0 < K; k0 += 32) {
#if HAVE_TDM
    if constexpr (USE_TDM) {
      if (wv == 0) {
        unsigned aoff = (unsigned)(unsigned long long)(const void*)&sAraw[0];
        unsigned boff = (unsigned)(unsigned long long)(const void*)&sBraw[0];
        // A tile: 32(x) x 64(y) from &A[m_blk*lda + k0], row stride lda
        tdm_load_2d(A + (size_t)m_blk * lda + k0, aoff, lda, 32, 64, 64);
        // B tile: 128(x) x 32(y) from &W[k0*N + n_blk], row stride N
        tdm_load_2d(W + (size_t)k0 * N + n_blk, boff, N, 128, 32, 32);
        __builtin_amdgcn_s_wait_tensorcnt(0);
      }
      __syncthreads();                   // landing zones visible to all waves
      // Convert A: f32 LDS -> f16 LDS (512 f4 chunks)
#pragma unroll
      for (int j = 0; j < 2; ++j) {
        int idx = tid + j * 256;
        int m = idx >> 3, k4 = (idx & 7) * 4;
        f4 av = *(const f4*)&sAraw[m * 32 + k4];
        h4 hv; hv[0] = (_Float16)av[0]; hv[1] = (_Float16)av[1];
        hv[2] = (_Float16)av[2]; hv[3] = (_Float16)av[3];
        *(h4*)&sA[m * 32 + k4] = hv;
      }
      // Convert+transpose B: sBraw[k][n] -> sB[n][k] (1024 f4 chunks)
#pragma unroll
      for (int j = 0; j < 4; ++j) {
        int idx = tid + j * 256;
        int k = idx >> 5, n4 = (idx & 31) * 4;
        f4 wq = *(const f4*)&sBraw[k * 128 + n4];
#pragma unroll
        for (int t = 0; t < 4; ++t)
          sB[(n4 + t) * 34 + k] = (_Float16)wq[t];
      }
    } else
#endif
    {
      // Phase 1: issue ALL tile loads (back-to-back global_load_b128).
      f4 areg[2], breg[4];
#pragma unroll
      for (int j = 0; j < 2; ++j) areg[j] = *(const f4*)(ap + j * lda32);
      if constexpr (!GUARD) {
#pragma unroll
        for (int j = 0; j < 4; ++j) breg[j] = *(const f4*)(wp + j * n8);
      }
      // Phase 2: convert + LDS store.
#pragma unroll
      for (int j = 0; j < 2; ++j) {
        int idx = tid + j * 256;
        int m = idx >> 3, k4 = (idx & 7) * 4;
        h4 hv; hv[0] = (_Float16)areg[j][0]; hv[1] = (_Float16)areg[j][1];
        hv[2] = (_Float16)areg[j][2]; hv[3] = (_Float16)areg[j][3];
        *(h4*)&sA[m * 32 + k4] = hv;
      }
      if constexpr (!GUARD) {
#pragma unroll
        for (int j = 0; j < 4; ++j) {
          int idx = tid + j * 256;
          int k = idx >> 5, n4 = (idx & 31) * 4;
#pragma unroll
          for (int t = 0; t < 4; ++t)
            sB[(n4 + t) * 34 + k] = (_Float16)breg[j][t];
        }
      } else {
        // Ragged N: clamp address, select 0 — loads stay pipelined.
#pragma unroll
        for (int j = 0; j < 16; ++j) {
          int idx = tid + j * 256;
          int k = idx >> 7, n = idx & 127;
          int gn = n_blk + n;
          int gnc = (gn < N - 1) ? gn : (N - 1);
          float v = W[(size_t)(k0 + k) * N + gnc];
          v = (gn < N) ? v : 0.f;
          sB[n * 34 + k] = (_Float16)v;
        }
      }
    }
    ap += 32;                 // advance K for next iteration
    wp += 4 * n8;             // += 32 rows of W
    __syncthreads();

    // ---------------- fragments + WMMA ----------------
    // A (16-bit 16x32): lane holds row M=lr; half-pair v carries
    // K = (v>=4?16:0) + 2*(v&3) + 8*g + sub.
    v16h af[2], bf[2];
#pragma unroll
    for (int mi = 0; mi < 2; ++mi) {
      int row = wm * 32 + mi * 16 + lr;
#pragma unroll
      for (int j = 0; j < 16; ++j) {
        int v = j >> 1, s = j & 1;
        int kk = ((v & 4) ? 16 : 0) + 2 * (v & 3) + 8 * g + s;
        af[mi][j] = sA[row * 32 + kk];
      }
    }
  // B (32x16): lane holds column N=lr; half j carries K = 16*(j>>3) + 8*g + (j&7).
#pragma unroll
    for (int ni = 0; ni < 2; ++ni) {
      int col = wn * 32 + ni * 16 + lr;
#pragma unroll
      for (int j = 0; j < 16; ++j)
        bf[ni][j] = sB[col * 34 + (16 * (j >> 3) + 8 * g + (j & 7))];
    }

#pragma unroll
    for (int mi = 0; mi < 2; ++mi)
#pragma unroll
      for (int ni = 0; ni < 2; ++ni)
        acc[mi][ni] = __builtin_amdgcn_wmma_f32_16x16x32_f16(
            false, af[mi], false, bf[ni], (short)0, acc[mi][ni],
            false, false);
    __syncthreads();
  }

  // Store. 32-bit C/D layout: lane -> N=lr, VGPR r -> M = r + 8*g.
  const int n_base = n_blk + wn * 32 + lr;
#pragma unroll
  for (int mi = 0; mi < 2; ++mi) {
    float* cp = C + (size_t)(m_blk + wm * 32 + mi * 16 + 8 * g) * N + n_base;
#pragma unroll
    for (int r = 0; r < 8; ++r) {
#pragma unroll
      for (int ni = 0; ni < 2; ++ni) {
        int gn = n_base + ni * 16;
        if (!GUARD || gn < N) {
          float v = acc[mi][ni][r];
          if (epilogue == 1) {              // softplus(v + bias)
            v += bias[gn];
            v = (v > 20.f) ? v : log1pf(__expf(v));
          }
          cp[ni * 16] = v;
        }
      }
      cp += N;
    }
  }
}

// Explicit instantiations so host/device passes always agree.
template __global__ void wmma_gemm_kernel<false, false>(
    const float*, const float*, float*, int, int, int, int, const float*, int);
template __global__ void wmma_gemm_kernel<true, false>(
    const float*, const float*, float*, int, int, int, int, const float*, int);
#if HAVE_TDM
template __global__ void wmma_gemm_kernel<false, true>(
    const float*, const float*, float*, int, int, int, int, const float*, int);
#endif

#if HAVE_TDM
#  define GEMM_FAST wmma_gemm_kernel<false, true>
#else
#  define GEMM_FAST wmma_gemm_kernel<false, false>
#endif

// ---------------------------------------------------------------------------
// Causal depthwise conv1d (k=4) + SiLU. xz is (B*L, 2*DI); x_val = cols 0..DI-1
// ---------------------------------------------------------------------------
__global__ __launch_bounds__(256) void conv_silu_kernel(
    const float* __restrict__ xz, const float* __restrict__ conv_w,
    const float* __restrict__ conv_b, float* __restrict__ xc)
{
  long i = (long)blockIdx.x * 256 + threadIdx.x;   // over B*L*DI
  int  d  = (int)(i % DIQ);
  long bl = i / DIQ;                               // b*L + l
  int  l  = (int)(bl % LQ);
  float acc = conv_b[d];
#pragma unroll
  for (int t = 0; t < 4; ++t) {
    int ll = l - 3 + t;
    if (ll >= 0)
      acc += xz[(bl - 3 + t) * (2 * DIQ) + d] * conv_w[d * 4 + t];
  }
  xc[i] = acc / (1.f + __expf(-acc));              // SiLU
}

// ---------------------------------------------------------------------------
// Selective scan: one thread per (b, d) channel, h[16] in registers,
// sequential over L.  Reads dt from dty, overwrites dty with y.
// ---------------------------------------------------------------------------
__global__ __launch_bounds__(256) void scan_kernel(
    const float* __restrict__ xc, const float* __restrict__ xdbl,
    const float* __restrict__ log_A_real, const float* __restrict__ D_param,
    float* __restrict__ dty)
{
  const int d = blockIdx.x * 256 + threadIdx.x;    // 0..DI-1
  const int b = blockIdx.y;

  float Arow[DSQ];
#pragma unroll
  for (int n = 0; n < DSQ; ++n) Arow[n] = -__expf(log_A_real[d * DSQ + n]);
  const float Dp = D_param[d];

  float h[DSQ];
#pragma unroll
  for (int n = 0; n < DSQ; ++n) h[n] = 0.f;

  for (int t = 0; t < LQ; ++t) {
    const long base = (long)b * LQ + t;
    const float dtv = dty[base * DIQ + d];         // softplus'd dt
    const float u   = xc[base * DIQ + d];
    const float du  = dtv * u;
    float y = Dp * u;

    const f4* bc = (const f4*)(xdbl + base * 96 + DTRQ);
    float Bv[DSQ], Cv[DSQ];
#pragma unroll
    for (int q = 0; q < 4; ++q) {
      f4 tb = bc[q], tc = bc[q + 4];
#pragma unroll
      for (int r = 0; r < 4; ++r) { Bv[4 * q + r] = tb[r]; Cv[4 * q + r] = tc[r]; }
    }
#pragma unroll
    for (int n = 0; n < DSQ; ++n) {
      h[n] = __expf(dtv * Arow[n]) * h[n] + du * Bv[n];
      y += Cv[n] * h[n];
    }
    dty[base * DIQ + d] = y;
  }
}

// ---------------------------------------------------------------------------
// Gate: y * silu(res), res = xz[:, DI:2*DI].  Output overwrites xc.
// ---------------------------------------------------------------------------
__global__ __launch_bounds__(256) void gate_kernel(
    const float* __restrict__ xz, const float* __restrict__ dty,
    float* __restrict__ out)
{
  long i = (long)blockIdx.x * 256 + threadIdx.x;   // over B*L*DI
  long row = i / DIQ;
  int  col = (int)(i % DIQ);
  float r = xz[row * (2 * DIQ) + DIQ + col];
  out[i] = dty[i] * (r / (1.f + __expf(-r)));
}

// ---------------------------------------------------------------------------
extern "C" void kernel_launch(void* const* d_in, const int* in_sizes, int n_in,
                              void* d_out, int out_size, void* d_ws, size_t ws_size,
                              hipStream_t stream)
{
  const float* x      = (const float*)d_in[0];
  const float* W_in   = (const float*)d_in[1];
  const float* conv_w = (const float*)d_in[2];
  const float* conv_b = (const float*)d_in[3];
  const float* W_x    = (const float*)d_in[4];
  const float* W_dt   = (const float*)d_in[5];
  const float* b_dt   = (const float*)d_in[6];
  const float* logA   = (const float*)d_in[7];
  const float* D_p    = (const float*)d_in[8];
  const float* W_out  = (const float*)d_in[9];
  float* out = (float*)d_out;

  const long Mrows = (long)BQ * LQ;                // 8192
  float* xz   = (float*)d_ws;                      // 8192 x 4096
  float* xc   = xz   + Mrows * 2 * DIQ;            // 8192 x 2048
  float* xdbl = xc   + Mrows * DIQ;                // 8192 x 96
  float* dty  = xdbl + Mrows * 96;                 // 8192 x 2048

  dim3 blk(256);
  const unsigned elem_blocks = (unsigned)((Mrows * DIQ) / 256);  // 65536

  // 1) in_proj: xz = x @ W_in                 (8192 x 4096, K=1024)
  GEMM_FAST<<<dim3((2 * DIQ) / 128, Mrows / 64), blk, 0, stream>>>(
      x, W_in, xz, (int)Mrows, 2 * DIQ, DMQ, DMQ, nullptr, 0);
  // 2) causal depthwise conv + SiLU -> xc
  conv_silu_kernel<<<dim3(elem_blocks), blk, 0, stream>>>(xz, conv_w, conv_b, xc);
  // 3) x_proj: xdbl = xc @ W_x                (8192 x 96, K=2048) — ragged N
  wmma_gemm_kernel<true, false><<<dim3(1, Mrows / 64), blk, 0, stream>>>(
      xc, W_x, xdbl, (int)Mrows, DTRQ + 2 * DSQ, DIQ, DIQ, nullptr, 0);
  // 4) dt = softplus(xdbl[:, :64] @ W_dt + b_dt)   (8192 x 2048, K=64, lda=96)
  GEMM_FAST<<<dim3(DIQ / 128, Mrows / 64), blk, 0, stream>>>(
      xdbl, W_dt, dty, (int)Mrows, DIQ, DTRQ, DTRQ + 2 * DSQ, b_dt, 1);
  // 5) selective scan (y overwrites dty)
  scan_kernel<<<dim3(DIQ / 256, BQ), blk, 0, stream>>>(xc, xdbl, logA, D_p, dty);
  // 6) gate: xc = y * silu(res)
  gate_kernel<<<dim3(elem_blocks), blk, 0, stream>>>(xz, dty, xc);
  // 7) out_proj: out = xc @ W_out             (8192 x 1024, K=2048)
  GEMM_FAST<<<dim3(DMQ / 128, Mrows / 64), blk, 0, stream>>>(
      xc, W_out, out, (int)Mrows, DMQ, DIQ, DIQ, nullptr, 0);
}
